// LocalSpectralTokenizer_11244224380986
// MI455X (gfx1250) — matrix operand order repaired and checked by reading, not compiled, not run on hardware
//
#include <hip/hip_runtime.h>
#include <cmath>

typedef __attribute__((ext_vector_type(16))) _Float16 v16h;
typedef __attribute__((ext_vector_type(8)))  float    v8f;

static constexpr int BATCH = 32;
static constexpr int CH    = 64;
static constexpr int LEN   = 8192;
static constexpr int WIN   = 32;
static constexpr int HOP   = 8;
static constexpr int NW    = 1021;   // (8192-32)/8 + 1 ; last start == LEN-WIN, no append
static constexpr int CAP   = 1026;   // max boundaries per batch
static constexpr int MS    = 1025;   // max segments per batch
static constexpr int MT    = 1100;   // max tokens per batch
static constexpr float PEN = 1.0f;   // PELT penalty
static constexpr double PI2 = 6.283185307179586476925286766559;

// ---------------------------------------------------------------- agg = mean over channels
__global__ void k_agg(const float* __restrict__ x, float* __restrict__ agg) {
    int idx = blockIdx.x * blockDim.x + threadIdx.x;
    if (idx >= BATCH * LEN) return;
    int b = idx / LEN, l = idx - b * LEN;
    const float* p = x + (size_t)b * CH * LEN + l;
    float s = 0.f;
    #pragma unroll 4
    for (int c = 0; c < CH; ++c) s += p[(size_t)c * LEN];
    agg[idx] = s * (1.0f / CH);
}

// ---------------------------------------------------------------- window features via WMMA DFT
// One wave per 16-window tile: frames(16x32) @ [cos|sin](32x16) with v_wmma_f32_16x16x32_f16.
__global__ __launch_bounds__(32) void k_winfeat(const float* __restrict__ agg,
                                                float* __restrict__ feat) {
    const int b    = blockIdx.y;
    const int m0   = blockIdx.x * 16;
    const int lane = threadIdx.x;
    const int hi   = lane >> 4;       // lane half
    const int row  = lane & 15;       // A-matrix row (window within tile)
    const int col  = lane & 15;       // B/C/D column (frequency-1)

    int win = m0 + row; if (win > NW - 1) win = NW - 1;   // clamp partial tile
    const float* ag = agg + (size_t)b * LEN + (size_t)win * HOP;

    v16h A, Bc, Bs;
    const float fo = (float)(col + 1);
    #pragma unroll
    for (int j = 0; j < 16; ++j) {
        // A layout (16-bit 16x32, ISA 7.12.2): v<4 -> K = 2v+h + hi*8 ; v>=4 -> K = 16 + 2(v-4)+h + hi*8
        int v = j >> 1, h = j & 1;
        int Ka = (v < 4) ? (2 * v + h + hi * 8) : (16 + 2 * (v - 4) + h + hi * 8);
        A[j] = (_Float16)ag[Ka];
        // B layout (16-bit 32x16): same K order as A: j<8 -> K = j + hi*8 ; j>=8 -> K = 16 + (j-8) + hi*8
        int Kb = (j < 8) ? (j + hi * 8) : (16 + (j - 8) + hi * 8);
        float ang = (float)(PI2 / 32.0) * fo * (float)Kb;
        Bc[j] = (_Float16)__cosf(ang);
        Bs[j] = (_Float16)__sinf(ang);
    }

    v8f cre = {};
    v8f cim = {};
    cre = __builtin_amdgcn_wmma_f32_16x16x32_f16(false, A, false, Bc, (short)0, cre, false, false);
    cim = __builtin_amdgcn_wmma_f32_16x16x32_f16(false, A, false, Bs, (short)0, cim, false, false);

    __shared__ float pt[16][17];   // [window row][freq-1], padded stride
    #pragma unroll
    for (int v = 0; v < 8; ++v) {
        int r = v + hi * 8;        // C/D layout: VGPR v holds M = v (+8 for lane>=16), N = lane&15
        pt[r][col] = cre[v] * cre[v] + cim[v] * cim[v];
    }
    __syncthreads();

    if (lane < 16) {
        int w = m0 + lane;
        if (w < NW) {
            double tot = 0.0, pf = 0.0, pmax = -1.0;
            int dom = 1;
            for (int c = 0; c < 16; ++c) {
                double p = (double)pt[lane][c];
                tot += p; pf += p * (double)(c + 1);
                if (p > pmax) { pmax = p; dom = c + 1; }
            }
            double t2 = tot < 1e-8 ? 1e-8 : tot;
            double cent = pf / t2;
            double var = 0.0;
            for (int c = 0; c < 16; ++c) {
                double df = (double)(c + 1) - cent;
                var += df * df * (double)pt[lane][c];
            }
            var /= t2;
            double bw = sqrt(var < 0.0 ? 0.0 : var) / 32.0;
            float* f = feat + ((size_t)b * NW + w) * 3;
            f[0] = (float)((double)dom / 32.0);
            f[1] = (float)bw;
            f[2] = (float)log1p(tot / 16.0);
        }
    }
}

// ---------------------------------------------------------------- PELT DP + backtrack + boundary filter
__global__ __launch_bounds__(1024) void k_dp(const float* __restrict__ feat,
                                             int* __restrict__ bnd, int* __restrict__ nbnd,
                                             int* __restrict__ tmp) {
    const int b = blockIdx.x, tid = threadIdx.x;
    __shared__ float preS[(NW + 1) * 3];
    __shared__ float preQ[(NW + 1) * 3];
    __shared__ float dpA[NW + 1];
    __shared__ int   prevA[NW + 1];
    __shared__ float redV[32];
    __shared__ int   redJ[32];

    if (tid == 0) {
        for (int d = 0; d < 3; ++d) { preS[d] = 0.f; preQ[d] = 0.f; }
        for (int i = 0; i < NW; ++i)
            for (int d = 0; d < 3; ++d) {
                float v = feat[((size_t)b * NW + i) * 3 + d];
                preS[(i + 1) * 3 + d] = preS[i * 3 + d] + v;
                preQ[(i + 1) * 3 + d] = preQ[i * 3 + d] + v * v;
            }
        dpA[0] = -PEN;
    }
    __syncthreads();

    for (int end = 1; end <= NW; ++end) {
        float best = 3.4e38f; int bj = 0x7fffffff;
        for (int j = tid; j < end; j += blockDim.x) {
            float sl = (float)(end - j);
            float sse = 0.f;
            #pragma unroll
            for (int d = 0; d < 3; ++d) {
                float ss = preS[end * 3 + d] - preS[j * 3 + d];
                float sq = preQ[end * 3 + d] - preQ[j * 3 + d];
                sse += sq - ss * ss / sl;
            }
            float cand = dpA[j] + sse + PEN;
            if (cand < best || (cand == best && j < bj)) { best = cand; bj = j; }
        }
        #pragma unroll
        for (int off = 16; off > 0; off >>= 1) {
            float ov = __shfl_down(best, off, 32);
            int   oj = __shfl_down(bj,   off, 32);
            if (ov < best || (ov == best && oj < bj)) { best = ov; bj = oj; }
        }
        if ((tid & 31) == 0) { redV[tid >> 5] = best; redJ[tid >> 5] = bj; }
        __syncthreads();
        if (tid == 0) {
            float bv = redV[0]; int bb = redJ[0];
            for (int wv = 1; wv < 32; ++wv)
                if (redV[wv] < bv || (redV[wv] == bv && redJ[wv] < bb)) { bv = redV[wv]; bb = redJ[wv]; }
            dpA[end] = bv; prevA[end] = bb;
        }
        __syncthreads();
    }

    if (tid == 0) {
        int cnt = 0, i = NW;
        while (i > 0 && cnt < CAP) { tmp[b * CAP + cnt++] = i; i = prevA[i]; }
        int nb = 0;
        bnd[b * CAP + nb++] = 0;
        for (int k = cnt - 1; k >= 1; --k) {           // wb[1:-1], ascending
            int w = tmp[b * CAP + k];
            if (w < 0 || w >= NW) continue;
            int t = HOP * w;
            int last = bnd[b * CAP + nb - 1];
            if (t <= last || t - last < 8 || LEN - t < 8) continue;
            if (nb < CAP - 1) bnd[b * CAP + nb++] = t;
        }
        bnd[b * CAP + nb++] = LEN;
        nbnd[b] = nb;
    }
}

// ---------------------------------------------------------------- per-segment DFT stats + patch length
__global__ __launch_bounds__(256) void k_seg(const float* __restrict__ agg,
                                             const int* __restrict__ bnd, const int* __restrict__ nbnd,
                                             int* __restrict__ segS, int* __restrict__ segE,
                                             int* __restrict__ segPL, float* __restrict__ segR,
                                             int* __restrict__ nseg) {
    const int b = blockIdx.x, si = blockIdx.y, tid = threadIdx.x;
    const int ns = nbnd[b] - 1;
    if (tid == 0 && si == 0) nseg[b] = ns;
    if (si >= ns) return;
    const int s = bnd[b * CAP + si], e = bnd[b * CAP + si + 1];
    const int n = e - s;
    const int nf = n / 2;
    const float* a = agg + (size_t)b * LEN + s;

    double pS = 0.0, pF = 0.0, pF2 = 0.0, pmax = -1.0;
    int dom = 0x7fffffff;
    for (int f = tid + 1; f <= nf; f += 256) {
        double re = 0.0, im = 0.0;
        double w0 = -PI2 * (double)f / (double)n;
        for (int k = 0; k < n; ++k) {
            double ang = w0 * (double)k, sv, cv;
            sincos(ang, &sv, &cv);
            double v = (double)a[k];
            re += v * cv; im += v * sv;
        }
        double p = re * re + im * im;
        pS += p; pF += p * (double)f; pF2 += p * (double)f * (double)f;
        if (p > pmax) { pmax = p; dom = f; }   // ascending f -> first occurrence
    }

    __shared__ double sS[256], sF[256], sF2[256], sM[256];
    __shared__ int sD[256];
    sS[tid] = pS; sF[tid] = pF; sF2[tid] = pF2; sM[tid] = pmax; sD[tid] = dom;
    __syncthreads();
    for (int off = 128; off > 0; off >>= 1) {
        if (tid < off) {
            sS[tid] += sS[tid + off];
            sF[tid] += sF[tid + off];
            sF2[tid] += sF2[tid + off];
            if (sM[tid + off] > sM[tid] || (sM[tid + off] == sM[tid] && sD[tid + off] < sD[tid])) {
                sM[tid] = sM[tid + off]; sD[tid] = sD[tid + off];
            }
        }
        __syncthreads();
    }
    if (tid == 0) {
        double tot = sS[0] < 1e-8 ? 1e-8 : sS[0];
        double cent = sF[0] / tot;
        double var = sF2[0] / tot - cent * cent;
        double bw = sqrt(var < 0.0 ? 0.0 : var) / (double)n;
        double dom_p = (double)n / (double)sD[0];
        double raw = dom_p / (1.0 + bw);               // ALPHA = BETA = 1
        int pl = (int)rint(raw / 2.0) * 2;             // round-half-even like Python round
        pl = pl < 8 ? 8 : (pl > 64 ? 64 : pl);
        segS[b * MS + si] = s;
        segE[b * MS + si] = e;
        segPL[b * MS + si] = pl;
        float* r = segR + ((size_t)b * MS + si) * 3;
        r[0] = (float)(dom_p / (double)LEN);
        r[1] = (float)bw;
        r[2] = (float)((double)n / (double)LEN);
    }
}

// ---------------------------------------------------------------- token enumeration + metadata outputs
__global__ void k_tokens(const int* __restrict__ segS, const int* __restrict__ segE,
                         const int* __restrict__ segPL, const float* __restrict__ segR,
                         const int* __restrict__ nseg,
                         int* __restrict__ tokS, int* __restrict__ tokE, int* __restrict__ tokSeg,
                         int* __restrict__ ntok, float* __restrict__ out, int mx) {
    int b = threadIdx.x;
    if (b >= BATCH) return;
    int nt = 0;
    int ns = nseg[b];
    for (int si = 0; si < ns; ++si) {
        int s = segS[b * MS + si], e = segE[b * MS + si], pl = segPL[b * MS + si];
        if (pl < 1) pl = 1;
        for (int aa = s; aa < e && nt < MT; aa += pl) {
            int z = aa + pl; if (z > e) z = e;
            if (z <= aa) continue;
            tokS[b * MT + nt] = aa; tokE[b * MT + nt] = z; tokSeg[b * MT + nt] = si; ++nt;
        }
    }
    ntok[b] = nt;

    const size_t NM = (size_t)BATCH * mx;
    float* maskO  = out + (size_t)BATCH * mx * (CH * 16);
    float* startO = maskO + NM;
    float* endO   = startO + NM;
    float* cenO   = endO + NM;
    float* spanO  = cenO + NM;
    float* regO   = spanO + NM;
    float* ntokO  = regO + NM * 3;
    for (int t = 0; t < mx; ++t) {
        size_t o = (size_t)b * mx + t;
        if (t < nt) {
            int s = tokS[b * MT + t], z = tokE[b * MT + t], si = tokSeg[b * MT + t];
            maskO[o]  = 1.0f;
            startO[o] = (float)s;
            endO[o]   = (float)z;
            cenO[o]   = (float)((double)(s + z - 1) * 0.5 / (double)(LEN - 1));
            spanO[o]  = (float)(z - s) / (float)LEN;
            const float* r = segR + ((size_t)b * MS + si) * 3;
            regO[o * 3 + 0] = r[0]; regO[o * 3 + 1] = r[1]; regO[o * 3 + 2] = r[2];
        } else {
            maskO[o] = 0.f; startO[o] = 0.f; endO[o] = 0.f; cenO[o] = 0.f; spanO[o] = 0.f;
            regO[o * 3 + 0] = 0.f; regO[o * 3 + 1] = 0.f; regO[o * 3 + 2] = 0.f;
        }
    }
    ntokO[b] = (float)nt;
}

// ---------------------------------------------------------------- patch gather (bilinear resize to 16 anchors)
__global__ __launch_bounds__(256) void k_gather(const float* __restrict__ x,
                                                const int* __restrict__ tokS, const int* __restrict__ tokE,
                                                const int* __restrict__ ntok,
                                                float* __restrict__ out, int mx) {
    const int bt = blockIdx.x;
    const int b = bt / mx, t = bt - b * mx;
    float* dst = out + (size_t)bt * (CH * 16);
    if (t >= ntok[b]) {
        for (int i = threadIdx.x; i < CH * 16; i += 256) dst[i] = 0.0f;
        return;
    }
    const int s = tokS[b * MT + t], z = tokE[b * MT + t];
    const int n = z - s;
    const float scale = (float)n / 16.0f;
    const float nm1 = (float)(n - 1);
    for (int i = threadIdx.x; i < CH * 16; i += 256) {
        int c = i >> 4, a = i & 15;
        float src = ((float)a + 0.5f) * scale - 0.5f;
        src = src < 0.f ? 0.f : (src > nm1 ? nm1 : src);
        float fl = floorf(src);
        int i0 = (int)fl;
        int i1 = i0 + 1; if (i1 > n - 1) i1 = n - 1;
        float w = src - fl;
        const float* xb = x + ((size_t)b * CH + c) * LEN + s;
        dst[i] = xb[i0] * (1.0f - w) + xb[i1] * w;
    }
}

// ----------------------------------------------------------------
extern "C" void kernel_launch(void* const* d_in, const int* in_sizes, int n_in,
                              void* d_out, int out_size, void* d_ws, size_t ws_size,
                              hipStream_t stream) {
    (void)in_sizes; (void)n_in; (void)ws_size;
    const float* x = (const float*)d_in[0];
    float* out = (float*)d_out;

    // out_size = B*mx*1032 + B  =>  recover mx
    int mx = 0;
    if (out_size > BATCH) mx = (out_size - BATCH) / (BATCH * 1032);
    if (mx <= 0) return;

    // workspace carve (256-byte aligned)
    char* ws = (char*)d_ws;
    size_t off = 0;
    auto carve = [&](size_t bytes) -> void* {
        void* p = ws + off;
        off = (off + bytes + 255) & ~(size_t)255;
        return p;
    };
    float* agg   = (float*)carve((size_t)BATCH * LEN * 4);
    float* feat  = (float*)carve((size_t)BATCH * NW * 3 * 4);
    int*   bnd   = (int*)  carve((size_t)BATCH * CAP * 4);
    int*   tmp   = (int*)  carve((size_t)BATCH * CAP * 4);
    int*   nbnd  = (int*)  carve((size_t)BATCH * 4);
    int*   segS  = (int*)  carve((size_t)BATCH * MS * 4);
    int*   segE  = (int*)  carve((size_t)BATCH * MS * 4);
    int*   segPL = (int*)  carve((size_t)BATCH * MS * 4);
    float* segR  = (float*)carve((size_t)BATCH * MS * 3 * 4);
    int*   nseg  = (int*)  carve((size_t)BATCH * 4);
    int*   tokS  = (int*)  carve((size_t)BATCH * MT * 4);
    int*   tokE  = (int*)  carve((size_t)BATCH * MT * 4);
    int*   tokSg = (int*)  carve((size_t)BATCH * MT * 4);
    int*   ntok  = (int*)  carve((size_t)BATCH * 4);

    // 1) channel mean
    k_agg<<<(BATCH * LEN + 255) / 256, 256, 0, stream>>>(x, agg);
    // 2) window spectral features via WMMA (64 tiles of 16 windows x 32 batches)
    k_winfeat<<<dim3((NW + 15) / 16, BATCH), 32, 0, stream>>>(agg, feat);
    // 3) PELT DP changepoints, one block per batch
    k_dp<<<BATCH, 1024, 0, stream>>>(feat, bnd, nbnd, tmp);
    // 4) per-segment spectral stats + patch length
    k_seg<<<dim3(BATCH, MS), 256, 0, stream>>>(agg, bnd, nbnd, segS, segE, segPL, segR, nseg);
    // 5) token enumeration + all metadata outputs (mask/start/end/center/span/regime/n_tokens)
    k_tokens<<<1, BATCH, 0, stream>>>(segS, segE, segPL, segR, nseg, tokS, tokE, tokSg, ntok, out, mx);
    // 6) patch gather with 16-anchor linear resize
    k_gather<<<BATCH * mx, 256, 0, stream>>>(x, tokS, tokE, ntok, out, mx);
}
